// DirectVoxGO_83322365542697
// MI455X (gfx1250) — matrix operand order repaired and checked by reading, not compiled, not run on hardware
//
#include <hip/hip_runtime.h>
#include <hip/hip_bf16.h>

#define M_SAMPLES 524288
#define NRAYS     8192
#define GX 160
#define GY 160
#define GZ 128
#define NFEAT 12
#define WIDTH 128
#define K0PAD 96                 // 75-dim input padded to 3 K-chunks of 32
#define KH    160                // 128 + 27 view-embed padded to 5 K-chunks of 32
#define ACT_SHIFT (-13.815509557964274f)

typedef __attribute__((ext_vector_type(16))) _Float16 v16h;
typedef __attribute__((ext_vector_type(8)))  float    v8f;

// ---------------------------------------------------------------- weights prep
__global__ void k_prepw(const float* __restrict__ W0, const float* __restrict__ W1,
                        const float* __restrict__ W2, const float* __restrict__ Wr,
                        _Float16* __restrict__ W0h, _Float16* __restrict__ W1h,
                        _Float16* __restrict__ W2h, _Float16* __restrict__ Wrh) {
  int idx = blockIdx.x * 256 + threadIdx.x;
  const int n0 = WIDTH * K0PAD;   // 12288
  const int n1 = WIDTH * WIDTH;   // 16384
  const int nr = 16 * KH;         // 2560
  if (idx < n0) {
    int n = idx / K0PAD, k = idx % K0PAD;           // store [n][k] (transposed)
    W0h[idx] = (_Float16)((k < 75) ? W0[k * WIDTH + n] : 0.f);
  } else if (idx < n0 + n1) {
    int j = idx - n0; int n = j / WIDTH, k = j % WIDTH;
    W1h[j] = (_Float16)W1[k * WIDTH + n];
  } else if (idx < n0 + 2 * n1) {
    int j = idx - n0 - n1; int n = j / WIDTH, k = j % WIDTH;
    W2h[j] = (_Float16)W2[k * WIDTH + n];
  } else if (idx < n0 + 2 * n1 + nr) {
    int j = idx - n0 - 2 * n1; int n = j / KH, k = j % KH;
    Wrh[j] = (_Float16)((n < 3 && k < 155) ? Wr[k * 3 + n] : 0.f);
  }
}

// ---------------------------------------------------------------- view posembed
__global__ void k_viewemb(const float* __restrict__ vd, float* __restrict__ ve) {
  int r = blockIdx.x * 256 + threadIdx.x;
  if (r >= NRAYS) return;
  float vv[3] = { vd[3 * r], vd[3 * r + 1], vd[3 * r + 2] };
  float* o = ve + (size_t)r * 27;
  o[0] = vv[0]; o[1] = vv[1]; o[2] = vv[2];
#pragma unroll
  for (int j = 0; j < 3; ++j)
#pragma unroll
    for (int p = 0; p < 4; ++p) {
      float xf = vv[j] * (float)(1 << p);
      o[3 + j * 4 + p]  = sinf(xf);
      o[15 + j * 4 + p] = cosf(xf);
    }
}

// ---------------------------------------------------------------- init
__global__ void k_init(int* __restrict__ rstart, float* __restrict__ ainv) {
  int r = blockIdx.x * 256 + threadIdx.x;
  if (r < NRAYS) { rstart[r] = -1; ainv[r] = 1.0f; }
}

// ---------------------------------------------------------------- per-sample
__global__ void k_sample(const float* __restrict__ xyz,
                         const unsigned char* __restrict__ mask,
                         const float* __restrict__ density,
                         const float* __restrict__ k0,
                         _Float16* __restrict__ h0,
                         float* __restrict__ alpha) {
  int i = blockIdx.x * 256 + threadIdx.x;
  if (i >= M_SAMPLES) return;
  float x = xyz[3 * i], y = xyz[3 * i + 1], z = xyz[3 * i + 2];

  // nearest-neighbor mask lookup: ijk = round(x*scale + shift), scale=shift=(G-1)/2
  float msx = (GX - 1) * 0.5f, msy = (GY - 1) * 0.5f, msz = (GZ - 1) * 0.5f;
  int mi = (int)roundf(x * msx + msx); mi = mi < 0 ? 0 : (mi > GX - 1 ? GX - 1 : mi);
  int mj = (int)roundf(y * msy + msy); mj = mj < 0 ? 0 : (mj > GY - 1 ? GY - 1 : mj);
  int mk = (int)roundf(z * msz + msz); mk = mk < 0 ? 0 : (mk > GZ - 1 ? GZ - 1 : mk);
  float m = mask[((size_t)mi * GY + mj) * GZ + mk] ? 1.f : 0.f;

  // trilinear setup (align_corners=True)
  float tx = fminf(fmaxf((x + 1.f) * 0.5f * (GX - 1), 0.f), (float)(GX - 1));
  float ty = fminf(fmaxf((y + 1.f) * 0.5f * (GY - 1), 0.f), (float)(GY - 1));
  float tz = fminf(fmaxf((z + 1.f) * 0.5f * (GZ - 1), 0.f), (float)(GZ - 1));
  int ix = (int)tx; if (ix > GX - 2) ix = GX - 2;
  int iy = (int)ty; if (iy > GY - 2) iy = GY - 2;
  int iz = (int)tz; if (iz > GZ - 2) iz = GZ - 2;
  float fx = tx - ix, fy = ty - iy, fz = tz - iz;

  float d = 0.f, feat[NFEAT];
#pragma unroll
  for (int c = 0; c < NFEAT; ++c) feat[c] = 0.f;
#pragma unroll
  for (int dx = 0; dx < 2; ++dx)
#pragma unroll
    for (int dy = 0; dy < 2; ++dy)
#pragma unroll
      for (int dz = 0; dz < 2; ++dz) {
        float w = (dx ? fx : 1.f - fx) * (dy ? fy : 1.f - fy) * (dz ? fz : 1.f - fz);
        size_t g = ((size_t)(ix + dx) * GY + (iy + dy)) * GZ + (iz + dz);
        d += w * density[g];
        const float* kp = k0 + g * NFEAT;
#pragma unroll
        for (int c = 0; c < NFEAT; ++c) feat[c] += w * kp[c];
      }

  // raw2alpha with interval 0.5:  1 - (1+exp(d+shift))^-0.5
  float a = (1.f - rsqrtf(1.f + expf(d + ACT_SHIFT))) * m;
  alpha[i] = a;

  // h0 row: [feat(12), xyz(3), sin(30), cos(30), pad(21)]  as fp16
  _Float16* hr = h0 + (size_t)i * K0PAD;
#pragma unroll
  for (int c = 0; c < NFEAT; ++c) hr[c] = (_Float16)feat[c];
  hr[12] = (_Float16)x; hr[13] = (_Float16)y; hr[14] = (_Float16)z;
  float pv[3] = { x, y, z };
#pragma unroll
  for (int j = 0; j < 3; ++j)
#pragma unroll
    for (int p = 0; p < 10; ++p) {
      float xf = pv[j] * (float)(1 << p);
      hr[15 + j * 10 + p] = (_Float16)sinf(xf);
      hr[45 + j * 10 + p] = (_Float16)cosf(xf);
    }
#pragma unroll
  for (int k = 75; k < K0PAD; ++k) hr[k] = (_Float16)0.f;
}

// ---------------------------------------------------------------- ray segmentation
__global__ void k_raystart(const int* __restrict__ ray_id, int* __restrict__ rstart) {
  int i = blockIdx.x * 256 + threadIdx.x;
  if (i >= M_SAMPLES) return;
  int r = ray_id[i];
  if (i == 0 || ray_id[i - 1] != r) rstart[r] = i;
}

__global__ void k_scan(const int* __restrict__ ray_id, const float* __restrict__ alpha,
                       const int* __restrict__ rstart, float* __restrict__ wgt,
                       float* __restrict__ ainv) {
  int r = blockIdx.x * 256 + threadIdx.x;
  if (r >= NRAYS) return;
  int s = rstart[r];
  if (s < 0) { ainv[r] = 1.f; return; }
  float T = 1.f;
  for (int i = s; i < M_SAMPLES && ray_id[i] == r; ++i) {
    float a = alpha[i];
    wgt[i] = T * a;
    T *= (1.f - a);
  }
  ainv[r] = T;
}

__global__ void k_outinit(const float* __restrict__ ainv, float* __restrict__ out) {
  int idx = blockIdx.x * 256 + threadIdx.x;
  if (idx < NRAYS * 3) out[idx] = ainv[idx / 3];   // white background term
}

// ---------------------------------------------------------------- WMMA fragments
// A (16x32 f16): lane half lh holds K pairs per ISA layout
__device__ __forceinline__ v16h load_a_pairs(const unsigned int* p, int kBase, int lh) {
  union { unsigned int u[8]; v16h v; } A;
#pragma unroll
  for (int r = 0; r < 8; ++r) {
    int k = kBase + ((r < 4) ? (lh * 8 + r * 2) : (16 + lh * 8 + (r - 4) * 2));
    A.u[r] = p[k >> 1];
  }
  return A.v;
}
// B (32x16 f16): lane = column, K pairs per ISA layout (same as A); p points at column start
__device__ __forceinline__ v16h load_b_pairs(const unsigned int* p, int kBase, int lh) {
  union { unsigned int u[8]; v16h v; } B;
#pragma unroll
  for (int r = 0; r < 8; ++r) {
    int k = kBase + ((r < 4) ? (lh * 8 + r * 2) : (16 + lh * 8 + (r - 4) * 2));
    B.u[r] = p[k >> 1];
  }
  return B.v;
}

template <int NK>
__device__ __forceinline__ void run_layer(const unsigned int* aRowDw,
                                          const _Float16* Wl, int Kpad,
                                          int r16, int lh, v8f acc[8]) {
#pragma unroll
  for (int c = 0; c < NK; ++c) {
    v16h a = load_a_pairs(aRowDw, c * 32, lh);
#pragma unroll
    for (int t = 0; t < 8; ++t) {
      const unsigned int* bp = (const unsigned int*)(Wl + (t * 16 + r16) * Kpad);
      v16h b = load_b_pairs(bp, c * 32, lh);
      acc[t] = __builtin_amdgcn_wmma_f32_16x16x32_f16(
          false, a, false, b, (short)0, acc[t], false, false);
    }
  }
}

// C/D 16x16 f32 layout: lane n = t*16 + (lane&15), row = v + 8*(lane>>4)
__device__ __forceinline__ void store_relu(_Float16* Hw, int stride, const float* bias,
                                           int r16, int lh, v8f acc[8]) {
#pragma unroll
  for (int t = 0; t < 8; ++t) {
    int n = t * 16 + r16;
    float bb = bias[n];
#pragma unroll
    for (int r = 0; r < 8; ++r) {
      float xv = acc[t][r] + bb;
      xv = xv > 0.f ? xv : 0.f;
      Hw[(r + lh * 8) * stride + n] = (_Float16)xv;
    }
  }
}

// --------- async L2 -> LDS weight staging (GLOBAL_LOAD_ASYNC_TO_LDS_B128, GVS mode)
__device__ __forceinline__ void stage_async(const _Float16* gsrc, _Float16* ldst,
                                            int n_b128, int tid) {
  // LDS aperture: low 32 bits of a generic LDS pointer == LDS byte address
  unsigned ldsBase = (unsigned)(unsigned long long)(void*)ldst;
  for (int j = tid; j < n_b128; j += 128) {
    unsigned voff = (unsigned)j * 16u;
    asm volatile("global_load_async_to_lds_b128 %0, %1, %2"
                 :: "v"(ldsBase + voff), "v"(voff), "s"(gsrc)
                 : "memory");
  }
  asm volatile("s_wait_asynccnt 0x0" ::: "memory");
}

// ---------------------------------------------------------------- MLP (WMMA)
__global__ __launch_bounds__(128) void k_mlp(
    const _Float16* __restrict__ h0,
    const _Float16* __restrict__ W0h, const _Float16* __restrict__ W1h,
    const _Float16* __restrict__ W2h, const _Float16* __restrict__ Wrh,
    const float* __restrict__ b0, const float* __restrict__ b1,
    const float* __restrict__ b2, const float* __restrict__ br,
    const float* __restrict__ wgt, const int* __restrict__ ray_id,
    const float* __restrict__ ve, float* __restrict__ out) {
  __shared__ _Float16 Wlds[WIDTH * WIDTH];   // 32 KB, one layer at a time
  __shared__ _Float16 Hlds[4][16 * KH];      // 20 KB, per-wave 16 x 160 tile

  const int tid  = threadIdx.x;
  const int lane = tid & 31, wave = tid >> 5;
  const int lh   = lane >> 4, r16 = lane & 15;
  const int rowBase = blockIdx.x * 64 + wave * 16;

  // ---- layer 0: W0 [128][96] fp16 -> LDS (async DMA)
  stage_async(W0h, Wlds, WIDTH * K0PAD * 2 / 16, tid);
  __syncthreads();
  {
    v8f acc[8] = {};
    const unsigned int* aRow =
        (const unsigned int*)(h0 + (size_t)(rowBase + r16) * K0PAD);
    run_layer<3>(aRow, Wlds, K0PAD, r16, lh, acc);
    store_relu(&Hlds[wave][0], KH, b0, r16, lh, acc);
  }
  __syncthreads();

  // ---- layer 1
  stage_async(W1h, Wlds, WIDTH * WIDTH * 2 / 16, tid);
  __syncthreads();
  {
    v8f acc[8] = {};
    const unsigned int* aRow = (const unsigned int*)(&Hlds[wave][r16 * KH]);
    run_layer<4>(aRow, Wlds, WIDTH, r16, lh, acc);
    store_relu(&Hlds[wave][0], KH, b1, r16, lh, acc);
  }
  __syncthreads();

  // ---- layer 2
  stage_async(W2h, Wlds, WIDTH * WIDTH * 2 / 16, tid);
  __syncthreads();
  {
    v8f acc[8] = {};
    const unsigned int* aRow = (const unsigned int*)(&Hlds[wave][r16 * KH]);
    run_layer<4>(aRow, Wlds, WIDTH, r16, lh, acc);
    store_relu(&Hlds[wave][0], KH, b2, r16, lh, acc);
  }
  __syncthreads();

  // ---- stage head weights Wr [16][160] fp16
  stage_async(Wrh, Wlds, 16 * KH * 2 / 16, tid);

  // ---- append view embedding to each row: cols 128..154 (+pad to 160)
  if (lane < 16) {
    int i = rowBase + r16;
    int ray = ray_id[i];
    const float* vep = ve + (size_t)ray * 27;
    _Float16* hr = &Hlds[wave][r16 * KH + WIDTH];
#pragma unroll
    for (int j = 0; j < 27; ++j) hr[j] = (_Float16)vep[j];
#pragma unroll
    for (int j = 27; j < 32; ++j) hr[j] = (_Float16)0.f;
  }
  __syncthreads();

  // ---- rgb head as WMMA: [16 rows x K=160] x [160 x 16 (3 valid)] -> 16x16
  v8f hacc = {};
  {
    const unsigned int* aRow = (const unsigned int*)(&Hlds[wave][r16 * KH]);
    const unsigned int* bp   = (const unsigned int*)(Wlds + r16 * KH);
#pragma unroll
    for (int c = 0; c < 5; ++c) {
      v16h a = load_a_pairs(aRow, c * 32, lh);
      v16h b = load_b_pairs(bp, c * 32, lh);
      hacc = __builtin_amdgcn_wmma_f32_16x16x32_f16(
          false, a, false, b, (short)0, hacc, false, false);
    }
  }

  // ---- sigmoid + weighted composite; D: column n = r16, rows v + 8*lh
  if (r16 < 3) {
    float bb = br[r16];
#pragma unroll
    for (int v = 0; v < 8; ++v) {
      int i = rowBase + v + lh * 8;
      float w = wgt[i];
      if (w != 0.f) {
        float rgb = 1.f / (1.f + expf(-(hacc[v] + bb)));
        atomicAdd(&out[ray_id[i] * 3 + r16], w * rgb);
      }
    }
  }
}

// ---------------------------------------------------------------- launch
extern "C" void kernel_launch(void* const* d_in, const int* in_sizes, int n_in,
                              void* d_out, int out_size, void* d_ws, size_t ws_size,
                              hipStream_t stream) {
  const float* xyz          = (const float*)d_in[0];
  const int*   ray_id       = (const int*)d_in[1];
  const float* viewdirs     = (const float*)d_in[2];
  const unsigned char* mask = (const unsigned char*)d_in[3];
  const float* density      = (const float*)d_in[4];
  const float* k0           = (const float*)d_in[5];
  const float* W0 = (const float*)d_in[6];
  const float* b0 = (const float*)d_in[7];
  const float* W1 = (const float*)d_in[8];
  const float* b1 = (const float*)d_in[9];
  const float* W2 = (const float*)d_in[10];
  const float* b2 = (const float*)d_in[11];
  const float* Wr = (const float*)d_in[12];
  const float* br = (const float*)d_in[13];
  float* out = (float*)d_out;
  (void)in_sizes; (void)n_in; (void)out_size; (void)ws_size;

  char* ws = (char*)d_ws;
  size_t off = 0;
  auto take = [&](size_t bytes) -> void* {
    void* p = ws + off;
    off = (off + bytes + 255) & ~(size_t)255;
    return p;
  };
  _Float16* h0  = (_Float16*)take((size_t)M_SAMPLES * K0PAD * 2);
  float* alpha  = (float*)take((size_t)M_SAMPLES * 4);
  float* wgt    = (float*)take((size_t)M_SAMPLES * 4);
  int*   rstart = (int*)take((size_t)NRAYS * 4);
  float* ainv   = (float*)take((size_t)NRAYS * 4);
  float* ve     = (float*)take((size_t)NRAYS * 27 * 4);
  _Float16* W0h = (_Float16*)take((size_t)WIDTH * K0PAD * 2);
  _Float16* W1h = (_Float16*)take((size_t)WIDTH * WIDTH * 2);
  _Float16* W2h = (_Float16*)take((size_t)WIDTH * WIDTH * 2);
  _Float16* Wrh = (_Float16*)take((size_t)16 * KH * 2);

  const int prep_total = WIDTH * K0PAD + 2 * WIDTH * WIDTH + 16 * KH;
  k_prepw   <<<(prep_total + 255) / 256, 256, 0, stream>>>(
      W0, W1, W2, Wr, W0h, W1h, W2h, Wrh);
  k_viewemb <<<NRAYS / 256, 256, 0, stream>>>(viewdirs, ve);
  k_init    <<<NRAYS / 256, 256, 0, stream>>>(rstart, ainv);
  k_sample  <<<M_SAMPLES / 256, 256, 0, stream>>>(xyz, mask, density, k0, h0, alpha);
  k_raystart<<<M_SAMPLES / 256, 256, 0, stream>>>(ray_id, rstart);
  k_scan    <<<NRAYS / 256, 256, 0, stream>>>(ray_id, alpha, rstart, wgt, ainv);
  k_outinit <<<(NRAYS * 3 + 255) / 256, 256, 0, stream>>>(ainv, out);
  k_mlp     <<<M_SAMPLES / 64, 128, 0, stream>>>(h0, W0h, W1h, W2h, Wrh,
                                                 b0, b1, b2, br,
                                                 wgt, ray_id, ve, out);
}
